// FlashAttentionWithPdrop_56573309224107
// MI455X (gfx1250) — matrix operand order repaired and checked by reading, not compiled, not run on hardware
//
#include <hip/hip_runtime.h>
#include <stdint.h>
#include <math.h>

// ---------------------------------------------------------------- types
typedef __attribute__((ext_vector_type(16))) __bf16 v16bf;
typedef __attribute__((ext_vector_type(8)))  __bf16 v8bf;
typedef __attribute__((ext_vector_type(8)))  float  v8f;

union Frag { v16bf v; v8bf h[2]; };

constexpr int Bb = 2, S = 2048, D = 2048, H = 16, Hd = 128;

static __device__ __forceinline__ uint16_t f2bf_bits(float f) {
  union { float f; uint32_t u; } c{f};
  uint32_t u = c.u;
  if ((u & 0x7fffffffu) > 0x7f800000u) return uint16_t((u >> 16) | 0x40); // qNaN
  return uint16_t((u + 0x7fffu + ((u >> 16) & 1u)) >> 16);                 // RNE
}
static __device__ __forceinline__ __bf16 f2bf(float f) {
  union { uint16_t u; __bf16 h; } c{f2bf_bits(f)};
  return c.h;
}

// Low 32 bits of a flat pointer into the LDS aperture are the LDS byte offset
// (ISA 10.2).
static __device__ __forceinline__ uint32_t lds_off32(const void* p) {
  return (uint32_t)(uintptr_t)p;
}

// GLOBAL_LOAD_ASYNC_TO_LDS_B128: 16B per lane, memory -> LDS, ASYNCcnt-tracked.
static __device__ __forceinline__ void async_cp16(uint32_t lds, const void* gsrc) {
  asm volatile("global_load_async_to_lds_b128 %0, %1, off"
               :: "v"(lds), "v"((uint64_t)(uintptr_t)gsrc) : "memory");
}
static __device__ __forceinline__ void wait_async(int pending) {
  if (pending) asm volatile("s_wait_asynccnt %0" :: "i"(4) : "memory");
  else         asm volatile("s_wait_asynccnt 0"  ::: "memory");
}
static __device__ __forceinline__ void wait_async2(int pending) {
  if (pending) asm volatile("s_wait_asynccnt %0" :: "i"(2) : "memory");
  else         asm volatile("s_wait_asynccnt 0"  ::: "memory");
}

// ---------------------------------------------------------------- fp32 -> bf16
__global__ void cvt_f32_bf16(const float4* __restrict__ in,
                             ushort4* __restrict__ out, int n4) {
  int i = blockIdx.x * blockDim.x + threadIdx.x;
  if (i < n4) {
    float4 f = in[i];
    ushort4 o;
    o.x = f2bf_bits(f.x); o.y = f2bf_bits(f.y);
    o.z = f2bf_bits(f.z); o.w = f2bf_bits(f.w);
    out[i] = o;
  }
}

// ---------------------------------------------------------------- WMMA GEMM
// C[M,N] = A[M,K] * W[N,K]^T + bias.  Block tile 128x128, wave tile 32x64.
// W K-tile (128 cols x 32 k) is double-buffered in LDS via async-to-LDS copies
// (shared by the 4 waves per column block); A fragments load direct from global.
// mode 0: bf16 out at [(b*H+h)*S+s]*Hd+hd          (Q,K head-major)
// mode 1: bf16 out at [(b*H+h)*Hd+hd]*S+s          (V transposed, K-major)
// mode 2: fp32 out at row*N+col                    (final projection)
constexpr int TBS = 40;   // padded LDS row stride (halves): 80B, 16B-aligned
__global__ __launch_bounds__(256)
void gemm_bf16_wmma(const __bf16* __restrict__ A, const __bf16* __restrict__ W,
                    const float* __restrict__ bias, float* __restrict__ outF,
                    __bf16* __restrict__ outB, int M, int N, int K, int mode) {
  const int lane = threadIdx.x & 31;
  const int w    = threadIdx.x >> 5;
  const int half = lane >> 4;
  const int l16  = lane & 15;
  const int t    = threadIdx.x;
  const int waveM = w & 3, waveN = w >> 2;               // 4 x 2 wave grid
  const int rowBase = blockIdx.y * 128 + waveM * 32;
  const int colBase0 = blockIdx.x * 128;

  __shared__ alignas(16) uint16_t tb[2][128 * TBS];

  // stage W tile [128 cols x 32 k] for k0 into tb[buf]: 512 x 16B chunks
  auto stage_w = [&](int k0, uint16_t* dst) {
#pragma unroll
    for (int i = 0; i < 2; ++i) {
      int ch  = t + i * 256;
      int row = ch >> 2;            // 0..127 (column of C = row of W)
      int sub = (ch & 3) * 8;       // halves
      const __bf16* g = W + (size_t)(colBase0 + row) * K + k0 + sub;
      async_cp16(lds_off32(dst + row * TBS + sub), g);
    }
  };

  const __bf16* pA[2];
#pragma unroll
  for (int mt = 0; mt < 2; ++mt)
    pA[mt] = A + (size_t)(rowBase + mt * 16 + l16) * K + 8 * half;

  const v8f zacc = {0.f,0.f,0.f,0.f,0.f,0.f,0.f,0.f};
  v8f acc[2][4];
#pragma unroll
  for (int mt = 0; mt < 2; ++mt)
#pragma unroll
    for (int nt = 0; nt < 4; ++nt) acc[mt][nt] = zacc;

  stage_w(0, tb[0]);

  for (int k0 = 0; k0 < K; k0 += 32) {
    const int buf = (k0 >> 5) & 1;
    const int more = (k0 + 32 < K);

    // A fragments direct from global (row = l16, K chunks {8h, 16+8h})
    Frag a[2];
#pragma unroll
    for (int mt = 0; mt < 2; ++mt) {
      a[mt].h[0] = *(const v8bf*)(pA[mt] + k0);
      a[mt].h[1] = *(const v8bf*)(pA[mt] + k0 + 16);
      __builtin_prefetch(pA[mt] + k0 + 32, 0, 1);   // speculative, unguarded
    }

    if (more) stage_w(k0 + 32, tb[buf ^ 1]);
    wait_async2(more);
    __syncthreads();                 // tb[buf] fully written by all waves

    Frag b[4];
#pragma unroll
    for (int nt = 0; nt < 4; ++nt) {
      const uint16_t* p = &tb[buf][(waveN * 64 + nt * 16 + l16) * TBS + 8 * half];
      b[nt].h[0] = *(const v8bf*)p;
      b[nt].h[1] = *(const v8bf*)(p + 16);
    }
#pragma unroll
    for (int mt = 0; mt < 2; ++mt)
#pragma unroll
      for (int nt = 0; nt < 4; ++nt)
        acc[mt][nt] = __builtin_amdgcn_wmma_f32_16x16x32_bf16(
            false, a[mt].v, false, b[nt].v, (short)0, acc[mt][nt], false, false);

    __syncthreads();                 // all reads of tb[buf] done before restage
  }

  // Epilogue: C layout -> lane(half,n) holds rows m = 8*half + r
#pragma unroll
  for (int mt = 0; mt < 2; ++mt)
#pragma unroll
    for (int nt = 0; nt < 4; ++nt)
#pragma unroll
      for (int r = 0; r < 8; ++r) {
        int row = rowBase + mt * 16 + 8 * half + r;
        int col = colBase0 + waveN * 64 + nt * 16 + l16;
        float v = acc[mt][nt][r] + bias[col];
        if (mode == 2) {
          outF[(size_t)row * N + col] = v;
        } else {
          int bb = row / S, s = row % S;
          int hh = col / Hd, hd = col % Hd;
          size_t idx = (mode == 0)
                           ? (((size_t)(bb * H + hh) * S + s) * Hd + hd)
                           : (((size_t)(bb * H + hh) * Hd + hd) * S + s);
          outB[idx] = f2bf(v);
        }
      }
}

// ---------------------------------------------------------------- flash attention
// q,k: bf16 [BH,S,Hd]  vt: bf16 [BH,Hd,S]  attn: bf16 [B*S, D]
// Block = (bh, 128-query tile); all 8 waves consume the SAME 32-key K/V tiles,
// so stage them once per block in double-buffered LDS via async-to-LDS copies.
constexpr int KTS = 136;  // K-tile LDS row stride (halves): 272B = 16*17
constexpr int VTS = 40;   // V-tile LDS row stride (halves): 80B  = 16*5
constexpr int PTS = 40;   // P staging row stride
__global__ __launch_bounds__(256)
void flash_attn_wmma(const __bf16* __restrict__ q, const __bf16* __restrict__ k,
                     const __bf16* __restrict__ vt, __bf16* __restrict__ attn) {
  const int lane = threadIdx.x & 31;
  const int w    = threadIdx.x >> 5;
  const int half = lane >> 4;
  const int l16  = lane & 15;
  const int t    = threadIdx.x;
  const int bh   = blockIdx.y;
  const int b    = bh / H, head = bh % H;
  const int qBase = blockIdx.x * 128 + w * 16;
  const float scale = 0.08838834764831845f; // 1/sqrt(128)

  const __bf16* qptr = q  + ((size_t)bh * S + qBase) * Hd;
  const __bf16* kptr = k  + (size_t)bh * S * Hd;
  const __bf16* vptr = vt + (size_t)bh * Hd * S;

  __shared__ alignas(16) uint16_t kt[2][32 * KTS];    // [key 0..31][d 0..127]
  __shared__ alignas(16) uint16_t vb_s[2][128 * VTS]; // [hd 0..127][kk 0..31]
  __shared__ alignas(16) uint16_t pb[8][16 * PTS];    // per-wave P staging

  // stage one 32-key K tile + V tile (4 async b128 per thread)
  auto stage_kv = [&](int j0, uint16_t* kd, uint16_t* vd) {
#pragma unroll
    for (int i = 0; i < 2; ++i) {                    // K: 512 x 16B chunks
      int ch = t + i * 256;
      int key = ch >> 4, sub = (ch & 15) * 8;
      async_cp16(lds_off32(kd + key * KTS + sub),
                 kptr + (size_t)(j0 + key) * Hd + sub);
    }
#pragma unroll
    for (int i = 0; i < 2; ++i) {                    // V: 512 x 16B chunks
      int ch = t + i * 256;
      int row = ch >> 2, sub = (ch & 3) * 8;
      async_cp16(lds_off32(vd + row * VTS + sub),
                 vptr + (size_t)row * S + j0 + sub);
    }
  };

  // Q fragments: 16 rows x 128 K, resident in registers
  Frag qa[4];
#pragma unroll
  for (int c = 0; c < 4; ++c) {
    const __bf16* p = qptr + (size_t)l16 * Hd + c * 32 + 8 * half;
    qa[c].h[0] = *(const v8bf*)p;
    qa[c].h[1] = *(const v8bf*)(p + 16);
  }

  const v8f zacc = {0.f,0.f,0.f,0.f,0.f,0.f,0.f,0.f};
  v8f o[8];
#pragma unroll
  for (int nt = 0; nt < 8; ++nt) o[nt] = zacc;
  float mrow[8], lrow[8];
#pragma unroll
  for (int r = 0; r < 8; ++r) { mrow[r] = -INFINITY; lrow[r] = 0.f; }

  stage_kv(0, kt[0], vb_s[0]);

  for (int j0 = 0; j0 < S; j0 += 32) {
    const int buf = (j0 >> 5) & 1;
    const int more = (j0 + 32 < S);
    if (more) stage_kv(j0 + 32, kt[buf ^ 1], vb_s[buf ^ 1]);
    wait_async(more);
    __syncthreads();                  // K/V tiles for j0 are resident

    // ---- scores: two 16x16 tiles over Hd=128 (8 WMMAs), K from LDS
    v8f s0 = zacc, s1 = zacc;
#pragma unroll
    for (int c = 0; c < 4; ++c) {
      Frag kb0, kb1;
      const uint16_t* p0 = &kt[buf][l16 * KTS + c * 32 + 8 * half];
      kb0.h[0] = *(const v8bf*)p0;  kb0.h[1] = *(const v8bf*)(p0 + 16);
      const uint16_t* p1 = &kt[buf][(16 + l16) * KTS + c * 32 + 8 * half];
      kb1.h[0] = *(const v8bf*)p1;  kb1.h[1] = *(const v8bf*)(p1 + 16);
      s0 = __builtin_amdgcn_wmma_f32_16x16x32_bf16(false, qa[c].v, false, kb0.v,
                                                   (short)0, s0, false, false);
      s1 = __builtin_amdgcn_wmma_f32_16x16x32_bf16(false, qa[c].v, false, kb1.v,
                                                   (short)0, s1, false, false);
    }

    // ---- online softmax; row m = 8*half + r lives on 16 lanes of one half
#pragma unroll
    for (int r = 0; r < 8; ++r) {
      float x0 = s0[r] * scale, x1 = s1[r] * scale;
      float mx = fmaxf(x0, x1);
#pragma unroll
      for (int off = 1; off < 16; off <<= 1)
        mx = fmaxf(mx, __shfl_xor(mx, off, 32));
      float mn = fmaxf(mrow[r], mx);
      float al = __expf(mrow[r] - mn);
      float p0 = __expf(x0 - mn), p1 = __expf(x1 - mn);
      float ps = p0 + p1;
#pragma unroll
      for (int off = 1; off < 16; off <<= 1)
        ps += __shfl_xor(ps, off, 32);
      lrow[r] = lrow[r] * al + ps;
      mrow[r] = mn;
#pragma unroll
      for (int nt = 0; nt < 8; ++nt) o[nt][r] *= al;
      int m = 8 * half + r;
      pb[w][m * PTS + l16]      = f2bf_bits(p0);
      pb[w][m * PTS + 16 + l16] = f2bf_bits(p1);
    }

    // ---- re-read P with A-fragment layout (layout transpose through LDS)
    Frag pa;
    const uint16_t* lp = &pb[w][l16 * PTS + 8 * half];
    pa.h[0] = *(const v8bf*)lp;
    pa.h[1] = *(const v8bf*)(lp + 16);

    // ---- O += P @ V, V fragments from LDS (K-major -> chunks {8h, 16+8h})
#pragma unroll
    for (int nt = 0; nt < 8; ++nt) {
      Frag vb;
      const uint16_t* p = &vb_s[buf][(nt * 16 + l16) * VTS + 8 * half];
      vb.h[0] = *(const v8bf*)p;
      vb.h[1] = *(const v8bf*)(p + 16);
      o[nt] = __builtin_amdgcn_wmma_f32_16x16x32_bf16(false, pa.v, false, vb.v,
                                                      (short)0, o[nt], false, false);
    }
    __syncthreads();                  // done reading buf before it is restaged
  }

  // ---- normalize and scatter to [B*S, D] (bf16) for the output projection
#pragma unroll
  for (int r = 0; r < 8; ++r) {
    float inv = 1.0f / lrow[r];
    int qi = qBase + 8 * half + r;
    __bf16* orow = attn + ((size_t)b * S + qi) * D + head * Hd;
#pragma unroll
    for (int nt = 0; nt < 8; ++nt)
      orow[nt * 16 + l16] = f2bf(o[nt][r] * inv);
  }
}

// ---------------------------------------------------------------- launcher
extern "C" void kernel_launch(void* const* d_in, const int* in_sizes, int n_in,
                              void* d_out, int out_size, void* d_ws, size_t ws_size,
                              hipStream_t stream) {
  (void)in_sizes; (void)n_in; (void)out_size; (void)ws_size;
  const float* x  = (const float*)d_in[0];
  const float* wq = (const float*)d_in[1];
  const float* bq = (const float*)d_in[2];
  const float* wk = (const float*)d_in[3];
  const float* bk = (const float*)d_in[4];
  const float* wv = (const float*)d_in[5];
  const float* bv = (const float*)d_in[6];
  const float* wo = (const float*)d_in[7];
  const float* bo = (const float*)d_in[8];
  float* out = (float*)d_out;

  const size_t M = (size_t)Bb * S;          // 4096
  unsigned char* ws = (unsigned char*)d_ws;
  __bf16* xb   = (__bf16*)ws;  ws += M * D * 2;                 // 16 MB
  __bf16* wqb  = (__bf16*)ws;  ws += (size_t)D * D * 2;         //  8 MB
  __bf16* wkb  = (__bf16*)ws;  ws += (size_t)D * D * 2;
  __bf16* wvb  = (__bf16*)ws;  ws += (size_t)D * D * 2;
  __bf16* wob  = (__bf16*)ws;  ws += (size_t)D * D * 2;
  __bf16* qb   = (__bf16*)ws;  ws += M * D * 2;                 // [BH,S,Hd]
  __bf16* kb   = (__bf16*)ws;  ws += M * D * 2;                 // [BH,S,Hd]
  __bf16* vtb  = (__bf16*)ws;  ws += M * D * 2;                 // [BH,Hd,S]
  __bf16* attb = (__bf16*)ws;  ws += M * D * 2;                 // [B*S, D]

  {
    int n4x = (int)(M * D / 4), n4w = (int)((size_t)D * D / 4);
    cvt_f32_bf16<<<(n4x + 255) / 256, 256, 0, stream>>>((const float4*)x,  (ushort4*)xb,  n4x);
    cvt_f32_bf16<<<(n4w + 255) / 256, 256, 0, stream>>>((const float4*)wq, (ushort4*)wqb, n4w);
    cvt_f32_bf16<<<(n4w + 255) / 256, 256, 0, stream>>>((const float4*)wk, (ushort4*)wkb, n4w);
    cvt_f32_bf16<<<(n4w + 255) / 256, 256, 0, stream>>>((const float4*)wv, (ushort4*)wvb, n4w);
    cvt_f32_bf16<<<(n4w + 255) / 256, 256, 0, stream>>>((const float4*)wo, (ushort4*)wob, n4w);
  }

  dim3 gg(D / 128, (unsigned)(M / 128)), blk(256);
  gemm_bf16_wmma<<<gg, blk, 0, stream>>>(xb, wqb, bq, nullptr, qb,  (int)M, D, D, 0);
  gemm_bf16_wmma<<<gg, blk, 0, stream>>>(xb, wkb, bk, nullptr, kb,  (int)M, D, D, 0);
  gemm_bf16_wmma<<<gg, blk, 0, stream>>>(xb, wvb, bv, nullptr, vtb, (int)M, D, D, 1);

  dim3 gf(S / 128, Bb * H);
  flash_attn_wmma<<<gf, blk, 0, stream>>>(qb, kb, vtb, attb);

  gemm_bf16_wmma<<<gg, blk, 0, stream>>>(attb, wob, bo, out, nullptr, (int)M, D, D, 2);
}
